// NTXentLoss_44624710205705
// MI455X (gfx1250) — matrix operand order repaired and checked by reading, not compiled, not run on hardware
//
#include <hip/hip_runtime.h>
#include <hip/hip_fp16.h>

// ---- CDNA5 vector types ----
typedef __attribute__((ext_vector_type(16))) _Float16 v16h;
typedef __attribute__((ext_vector_type(8)))  _Float16 v8h;
typedef __attribute__((ext_vector_type(8)))  float    v8f;

#define NB     4096      // B
#define DD     512       // D
#define TB     8192      // 2B
#define INV_T  5.0f      // 1 / 0.2
#define EPS_N  1e-12f

union V16U { v16h v; v8h h[2]; };

// ------------------------------------------------------------------
// Kernel 1: L2-normalize each of the 8192 rows (zis rows 0..B-1,
// zjs rows B..2B-1), write f16 repsH[row][512]. One wave per row.
// ------------------------------------------------------------------
__global__ void k_normalize(const float* __restrict__ zis,
                            const float* __restrict__ zjs,
                            _Float16* __restrict__ repsH) {
    const int wave = threadIdx.x >> 5;
    const int lane = threadIdx.x & 31;
    const int row  = blockIdx.x * 8 + wave;           // grid = 1024 blocks
    const float* src = (row < NB) ? (zis + (size_t)row * DD)
                                  : (zjs + (size_t)(row - NB) * DD);

    // each lane owns 16 consecutive floats
    const float4* s4 = (const float4*)(src + lane * 16);
    float4 x0 = s4[0], x1 = s4[1], x2 = s4[2], x3 = s4[3];
    float v[16] = {x0.x,x0.y,x0.z,x0.w, x1.x,x1.y,x1.z,x1.w,
                   x2.x,x2.y,x2.z,x2.w, x3.x,x3.y,x3.z,x3.w};

    float ss = 0.f;
#pragma unroll
    for (int t = 0; t < 16; ++t) ss += v[t] * v[t];
#pragma unroll
    for (int m = 16; m >= 1; m >>= 1) ss += __shfl_xor(ss, m, 32);

    const float scale = 1.0f / fmaxf(sqrtf(ss), EPS_N);

    v8h o0, o1;
#pragma unroll
    for (int t = 0; t < 8; ++t) {
        o0[t] = (_Float16)(v[t]     * scale);
        o1[t] = (_Float16)(v[t + 8] * scale);
    }
    v8h* dst = (v8h*)(repsH + (size_t)row * DD + lane * 16);
    dst[0] = o0;
    dst[1] = o1;
}

// ------------------------------------------------------------------
// Kernel 2: positives: pos[i] = dot(repsH[i], repsH[i+B]) in f32.
// One wave per i.
// ------------------------------------------------------------------
__global__ void k_positives(const _Float16* __restrict__ repsH,
                            float* __restrict__ pos) {
    const int wave = threadIdx.x >> 5;
    const int lane = threadIdx.x & 31;
    const int i    = blockIdx.x * 8 + wave;           // grid = 512 blocks

    const v8h* pa = (const v8h*)(repsH + (size_t)i        * DD + lane * 16);
    const v8h* pb = (const v8h*)(repsH + (size_t)(i + NB) * DD + lane * 16);
    v8h a0 = pa[0], a1 = pa[1], b0 = pb[0], b1 = pb[1];

    float acc = 0.f;
#pragma unroll
    for (int t = 0; t < 8; ++t) {
        acc += (float)a0[t] * (float)b0[t];
        acc += (float)a1[t] * (float)b1[t];
    }
#pragma unroll
    for (int m = 16; m >= 1; m >>= 1) acc += __shfl_xor(acc, m, 32);
    if (lane == 0) pos[i] = acc;
}

// ------------------------------------------------------------------
// Kernel 3: fused sim GEMM + exp-sum per row.
// Block = 16-row tile (grid 512), 8 waves; wave w scans column tiles
// [w*64, w*64+64). A-block register resident (128 VGPRs, protected
// from spilling by launch_bounds). B fragments are TRIPLE-buffered
// (prefetch distance 2): the fragment consumed by each WMMA was
// issued two steps earlier, and the first two fragments of tile t+1
// are issued during the last two steps of tile t, so no WMMA ever
// needs a full loadcnt drain.
// ------------------------------------------------------------------
__global__ void __launch_bounds__(256, 1)
k_simloss(const _Float16* __restrict__ repsH,
          const float* __restrict__ pos,
          float* __restrict__ rowloss) {
    __shared__ float sdata[8][16];

    const int wave = threadIdx.x >> 5;
    const int lane = threadIdx.x & 31;
    const int m0   = blockIdx.x * 16;
    const int half = lane >> 4;           // 0 or 1

    // ---- load A fragments for this 16-row block (contiguous K order, matches B)
    // lane m=lane&15; lanes 0-15 hold K {0..15}, lanes 16-31 {16..31}
    const int arow  = m0 + (lane & 15);
    const int abase = half << 4;          // 0 or 16 halves
    const _Float16* aptr = repsH + (size_t)arow * DD;

    V16U afrag[16];
#pragma unroll
    for (int kk = 0; kk < 16; ++kk) {
        const int k0 = kk * 32;
        afrag[kk].h[0] = *(const v8h*)(aptr + k0 + abase);
        afrag[kk].h[1] = *(const v8h*)(aptr + k0 + abase + 8);
    }

    // ---- B base for this lane: col n = n0 + (lane&15); lanes 0-15 hold
    // K {0..15}, lanes 16-31 hold K {16..31} of each 32-deep step
    const int  bcol_in_tile = lane & 15;
    const int  bkoff        = half << 4;  // 0 or 16 halves

    float s[8] = {0.f,0.f,0.f,0.f,0.f,0.f,0.f,0.f};

    const int tile0 = wave * 64;
    const _Float16* bbase =
        repsH + (size_t)(tile0 * 16 + bcol_in_tile) * DD + bkoff;

    // prime the pipeline: fragments 0 and 1 of tile 0
    V16U bc, bn;
    bc.h[0] = ((const v8h*)bbase)[0];
    bc.h[1] = ((const v8h*)bbase)[1];
    bn.h[0] = ((const v8h*)(bbase + 32))[0];
    bn.h[1] = ((const v8h*)(bbase + 32))[1];

    for (int t = 0; t < 64; ++t) {
        const int n0 = (tile0 + t) * 16;
        // next tile base (clamped on the last tile: re-reads current tile,
        // values are discarded, never reads past repsH)
        const _Float16* bnext_tile = bbase + ((t < 63) ? (size_t)16 * DD : 0);

        v8f acc = {};
#pragma unroll
        for (int kk = 0; kk < 16; ++kk) {
            // prefetch distance 2: fragment kk+2 (rolls into next tile)
            const _Float16* pf = (kk < 14) ? (bbase + (kk + 2) * 32)
                                           : (bnext_tile + (kk - 14) * 32);
            V16U b2;
            b2.h[0] = ((const v8h*)pf)[0];
            b2.h[1] = ((const v8h*)pf)[1];

            acc = __builtin_amdgcn_wmma_f32_16x16x32_f16(
                    false, afrag[kk].v, false, bc.v,
                    (short)0, acc, false, false);
            bc = bn;
            bn = b2;
        }

        // per-slot exp accumulation. n0/m0 are wave-uniform, so the
        // diagonal-containing tile check is a scalar branch: the hot path
        // has no per-lane compares.
        if (n0 != m0) {
#pragma unroll
            for (int r = 0; r < 8; ++r)
                s[r] += __expf(acc[r] * INV_T);
        } else {
            const int col = n0 + (lane & 15);
#pragma unroll
            for (int r = 0; r < 8; ++r) {
                const int row = m0 + r + (half << 3);
                const float e = __expf(acc[r] * INV_T);
                s[r] += (row == col) ? 0.0f : e;
            }
        }

        bbase = bnext_tile;
    }

    // ---- reduce the 16 column-slots of each row (within 16-lane halves)
#pragma unroll
    for (int r = 0; r < 8; ++r) {
        float t = s[r];
        t += __shfl_xor(t, 1, 16);
        t += __shfl_xor(t, 2, 16);
        t += __shfl_xor(t, 4, 16);
        t += __shfl_xor(t, 8, 16);
        if ((lane & 15) == 0) sdata[wave][r + (half << 3)] = t;
    }
    __syncthreads();

    // ---- combine the 8 wave partials in fixed order (deterministic),
    // fold in the positive logit, emit per-row loss
    if (threadIdx.x < 16) {
        float sum = 0.f;
#pragma unroll
        for (int w = 0; w < 8; ++w) sum += sdata[w][threadIdx.x];
        const int grow = m0 + threadIdx.x;
        const int i    = (grow < NB) ? grow : grow - NB;
        const float p  = pos[i] * INV_T;
        rowloss[grow]  = logf(sum + __expf(p)) - p;
    }
}

// ------------------------------------------------------------------
// Kernel 4: deterministic mean of 8192 row losses -> d_out[0].
// Single block of 256 threads.
// ------------------------------------------------------------------
__global__ void k_reduce(const float* __restrict__ rowloss,
                         float* __restrict__ out) {
    __shared__ float sw[8];
    const int tid  = threadIdx.x;
    const int lane = tid & 31;
    const int wave = tid >> 5;

    float acc = 0.f;
#pragma unroll
    for (int j = 0; j < TB / 256; ++j) acc += rowloss[j * 256 + tid];
#pragma unroll
    for (int m = 16; m >= 1; m >>= 1) acc += __shfl_xor(acc, m, 32);
    if (lane == 0) sw[wave] = acc;
    __syncthreads();
    if (tid == 0) {
        float total = 0.f;
#pragma unroll
        for (int w = 0; w < 8; ++w) total += sw[w];
        out[0] = total / (float)TB;
    }
}

// ------------------------------------------------------------------
extern "C" void kernel_launch(void* const* d_in, const int* in_sizes, int n_in,
                              void* d_out, int out_size, void* d_ws, size_t ws_size,
                              hipStream_t stream) {
    const float* zis = (const float*)d_in[0];
    const float* zjs = (const float*)d_in[1];
    float* out = (float*)d_out;

    // workspace layout
    const size_t REPS_BYTES = (size_t)TB * DD * sizeof(_Float16); // 8 MB
    const size_t POS_BYTES  = (size_t)NB * sizeof(float);         // 16 KB
    _Float16* repsH   = (_Float16*)d_ws;
    float*    pos     = (float*)((char*)d_ws + REPS_BYTES);
    float*    rowloss = (float*)((char*)d_ws + REPS_BYTES + POS_BYTES);

    k_normalize<<<dim3(TB / 8), dim3(256), 0, stream>>>(zis, zjs, repsH);
    k_positives<<<dim3(NB / 8), dim3(256), 0, stream>>>(repsH, pos);
    k_simloss  <<<dim3(TB / 16), dim3(256), 0, stream>>>(repsH, pos, rowloss);
    k_reduce   <<<dim3(1),       dim3(256), 0, stream>>>(rowloss, out);
}
